// SLDS_CANN_VAE_35888746725395
// MI455X (gfx1250) — matrix operand order repaired and checked by reading, not compiled, not run on hardware
//
#include <hip/hip_runtime.h>
#include <hip/hip_bf16.h>
#include <math.h>

// ---------------------------------------------------------------------------
// Problem dims (match reference)
// ---------------------------------------------------------------------------
#define Bsz 32
#define Tsz 1024
#define Dsz 128
#define Lsz 64
#define Kcat 8
#define Nsz 1024
#define Hsz 1024
#define DT_STEP 0.1f
#define NWG_REC 64           // persistent workgroups for the recurrences

typedef __attribute__((ext_vector_type(16))) __bf16 v16bf;
typedef __attribute__((ext_vector_type(8)))  float  v8f;

union FragU { uint4 q[2]; v16bf v; };

// A-fragment (16x32 bf16): lane = M (lane&15); halves 0..7 = K s..s+7,
// halves 8..15 = K s+16..s+23, s = (lane&16)?8:0.  Two 16B loads per lane.
__device__ __forceinline__ v16bf load_a_frag(const __bf16* __restrict__ p,
                                             long ld, int row0, long k0) {
  const int lane = threadIdx.x & 31;
  const __bf16* r = p + (long)(row0 + (lane & 15)) * ld + k0 + ((lane & 16) ? 8 : 0);
  FragU f;
  f.q[0] = *(const uint4*)(r);
  f.q[1] = *(const uint4*)(r + 16);
  return f.v;
}

// B-fragment (32x16 bf16): lane holds column n = n0+(lane&15); halves 0..7 =
// K s..s+7, halves 8..15 = K s+16..s+23, s = (lane&16)?8:0.  Two 16B loads per lane.
// Since gates = h @ W^T, B[k][n] = W[n][k]: row n of W is contiguous in k.
__device__ __forceinline__ v16bf load_b_frag(const __bf16* __restrict__ w,
                                             long ld, int n0, int k0) {
  const int lane = threadIdx.x & 31;
  const __bf16* r = w + (long)(n0 + (lane & 15)) * ld + k0 + ((lane & 16) ? 8 : 0);
  FragU f;
  f.q[0] = *(const uint4*)(r);
  f.q[1] = *(const uint4*)(r + 16);
  return f.v;
}

__device__ __forceinline__ v8f wmma_bf16(v16bf a, v16bf b, v8f c) {
  return __builtin_amdgcn_wmma_f32_16x16x32_bf16(false, a, false, b, (short)0, c,
                                                 false, false);
}

__device__ __forceinline__ float sigmoidf(float x) { return 1.f / (1.f + expf(-x)); }

// ---------------------------------------------------------------------------
// Device-wide generation barrier (persistent-kernel step sync).
// Standard pattern: every thread fences, block syncs, thread0 arrives with
// agent-scope atomics; last arriver resets the counter and bumps `gen`.
// ---------------------------------------------------------------------------
__device__ __forceinline__ void grid_barrier(unsigned* cnt, unsigned* gen,
                                             unsigned nwg, unsigned target) {
  __threadfence();
  __syncthreads();
  if (threadIdx.x == 0) {
    unsigned prev = __hip_atomic_fetch_add(cnt, 1u, __ATOMIC_ACQ_REL,
                                           __HIP_MEMORY_SCOPE_AGENT);
    if (prev == nwg - 1) {
      __hip_atomic_store(cnt, 0u, __ATOMIC_RELAXED, __HIP_MEMORY_SCOPE_AGENT);
      __hip_atomic_fetch_add(gen, 1u, __ATOMIC_ACQ_REL, __HIP_MEMORY_SCOPE_AGENT);
    } else {
      while (__hip_atomic_load(gen, __ATOMIC_ACQUIRE, __HIP_MEMORY_SCOPE_AGENT) <
             target) {
        __builtin_amdgcn_s_sleep(1);
      }
    }
  }
  __syncthreads();
  __threadfence();
}

// ---------------------------------------------------------------------------
// Elementwise helpers
// ---------------------------------------------------------------------------
__global__ void f2bf_kernel(const float* __restrict__ src, __bf16* __restrict__ dst,
                            long n) {
  long i = (long)blockIdx.x * blockDim.x + threadIdx.x;
  if (i < n) dst[i] = (__bf16)src[i];
}

// Wk is (8, H); pad to 16 rows of zeros so a full B-fragment stays in bounds.
__global__ void wk_pad_kernel(const float* __restrict__ Wk, __bf16* __restrict__ dst) {
  long i = (long)blockIdx.x * blockDim.x + threadIdx.x;
  if (i < 16L * Hsz) {
    int row = (int)(i >> 10);
    dst[i] = (row < Kcat) ? (__bf16)Wk[(long)row * Hsz + (i & 1023)] : (__bf16)0.f;
  }
}

__global__ void init_barrier_kernel(unsigned* bar) {
  if (threadIdx.x < 8) bar[threadIdx.x] = 0u;
}

// ---------------------------------------------------------------------------
// Persistent LSTM scan.  64 WGs x 256 threads; WG wg owns hidden units
// [wg*16, wg*16+16).  LDS holds its 64 W_hh rows (i,f,g,o x 16) = 128KB bf16
// plus the matching 64 W_ih rows.  Wave w computes gate tile (w&3) for
// batch-tile (w>>2): 4 + 32 WMMAs per wave per step.  Gate fusion keeps c in
// registers; h ping-pongs through L2 with a grid barrier per step.
// ---------------------------------------------------------------------------
__global__ void lstm_scan_kernel(const __bf16* __restrict__ ybf,   // (B,T,D)
                                 const __bf16* __restrict__ Wih,   // (4H,D)
                                 const __bf16* __restrict__ Whh,   // (4H,H)
                                 const float* __restrict__ b_ih,
                                 const float* __restrict__ b_hh,
                                 __bf16* __restrict__ hbuf,        // 2*B*H
                                 __bf16* __restrict__ hseq,        // (B,T,H)
                                 unsigned* cnt, unsigned* gen) {
  extern __shared__ char smem[];
  __bf16* sWhh  = (__bf16*)smem;                   // 64*1024 bf16
  __bf16* sWih  = sWhh + 64 * Hsz;                 // 64*128 bf16
  float*  sAcc  = (float*)(sWih + 64 * Dsz);       // 4*32*16 f32
  float*  sBias = sAcc + 4 * Bsz * 16;             // 64 f32

  const int wg = blockIdx.x;
  const int tid = threadIdx.x;
  const int wave = tid >> 5, lane = tid & 31;
  const int gate = wave & 3, mt = wave >> 2;
  const int m0 = mt * 16, j0 = wg * 16;

  { // stage weight slices into LDS (once)
    const uint4* s4 = (const uint4*)Whh;           // row = H/8 = 128 uint4
    uint4* d4 = (uint4*)sWhh;
    for (int i = tid; i < 64 * (Hsz / 8); i += blockDim.x) {
      int row = i >> 7, ch = i & 127, g = row >> 4, r = row & 15;
      d4[i] = s4[(long)(g * Hsz + j0 + r) * (Hsz / 8) + ch];
    }
    const uint4* s2 = (const uint4*)Wih;           // row = D/8 = 16 uint4
    uint4* d2 = (uint4*)sWih;
    for (int i = tid; i < 64 * (Dsz / 8); i += blockDim.x) {
      int row = i >> 4, ch = i & 15, g = row >> 4, r = row & 15;
      d2[i] = s2[(long)(g * Hsz + j0 + r) * (Dsz / 8) + ch];
    }
    for (int i = tid; i < 64; i += blockDim.x) {
      int g = i >> 4, n = i & 15;
      sBias[i] = b_ih[g * Hsz + j0 + n] + b_hh[g * Hsz + j0 + n];
    }
  }
  // h0 = 0 (each WG zeroes its own columns)
  for (int i = tid; i < Bsz * 16; i += blockDim.x)
    hbuf[(long)(i >> 4) * Hsz + j0 + (i & 15)] = (__bf16)0.f;
  grid_barrier(cnt, gen, NWG_REC, 1u);

  float cst[2] = {0.f, 0.f};

  for (int t = 0; t < Tsz; ++t) {
    const __bf16* hin = hbuf + (long)(t & 1) * Bsz * Hsz;
    __bf16* hout      = hbuf + (long)((t + 1) & 1) * Bsz * Hsz;

    v8f acc = {};
    for (int k = 0; k < Dsz; k += 32) {            // x_t @ W_ih^T slice
      v16bf a = load_a_frag(ybf, (long)Tsz * Dsz, m0, (long)t * Dsz + k);
      acc = wmma_bf16(a, load_b_frag(sWih, Dsz, gate * 16, k), acc);
    }
    for (int k = 0; k < Hsz; k += 32) {            // h @ W_hh^T slice
      v16bf a = load_a_frag(hin, Hsz, m0, k);
      acc = wmma_bf16(a, load_b_frag(sWhh, Hsz, gate * 16, k), acc);
    }
#pragma unroll
    for (int v = 0; v < 8; ++v) {
      int m = m0 + v + ((lane & 16) ? 8 : 0);
      sAcc[(gate * Bsz + m) * 16 + (lane & 15)] = acc[v];
    }
    __syncthreads();
#pragma unroll
    for (int pi = 0; pi < 2; ++pi) {               // 512 (batch,unit) pairs
      int p = tid + pi * 256;
      int m = p >> 4, n = p & 15;
      float gi = sAcc[(0 * Bsz + m) * 16 + n] + sBias[0 * 16 + n];
      float gf = sAcc[(1 * Bsz + m) * 16 + n] + sBias[1 * 16 + n];
      float gg = sAcc[(2 * Bsz + m) * 16 + n] + sBias[2 * 16 + n];
      float go = sAcc[(3 * Bsz + m) * 16 + n] + sBias[3 * 16 + n];
      float c = sigmoidf(gf) * cst[pi] + sigmoidf(gi) * tanhf(gg);
      float h = sigmoidf(go) * tanhf(c);
      cst[pi] = c;
      __bf16 hb = (__bf16)h;
      hout[(long)m * Hsz + j0 + n] = hb;
      hseq[((long)m * Tsz + t) * Hsz + j0 + n] = hb;
    }
    grid_barrier(cnt, gen, NWG_REC, (unsigned)(t + 2));
  }
}

// ---------------------------------------------------------------------------
// Persistent CANN scan: u += DT*(-u + relu(u)@W_rec^T + I_t).
// 64 WGs x 256 threads; WG owns 16 units; W_rec slice (16x1024 bf16 = 32KB)
// lives in LDS.  Wave w: batch-tile w&1, K-slice w>>1 (8 WMMAs/step),
// partials reduced through LDS.  u in registers, relu(u) ping-pongs bf16.
// ---------------------------------------------------------------------------
__global__ void cann_scan_kernel(const __bf16* __restrict__ Wrec,  // (N,N)
                                 const float* __restrict__ Iext,   // (B,T,N)
                                 __bf16* __restrict__ rbuf,        // 2*B*N
                                 __bf16* __restrict__ useq,        // (B,T,N)
                                 unsigned* cnt, unsigned* gen) {
  extern __shared__ char smem[];
  __bf16* sW = (__bf16*)smem;                      // 16*1024 bf16
  float* sAcc = (float*)(sW + 16 * Nsz);           // 8*16*16 f32

  const int wg = blockIdx.x, tid = threadIdx.x;
  const int wave = tid >> 5, lane = tid & 31;
  const int mt = wave & 1, ks = wave >> 1;
  const int m0 = mt * 16, j0 = wg * 16;

  {
    const uint4* s4 = (const uint4*)Wrec;
    uint4* d4 = (uint4*)sW;
    for (int i = tid; i < 16 * (Nsz / 8); i += blockDim.x)
      d4[i] = s4[(long)(j0 + (i >> 7)) * (Nsz / 8) + (i & 127)];
  }
  for (int i = tid; i < Bsz * 16; i += blockDim.x)   // relu(u0) = 0
    rbuf[(long)(i >> 4) * Nsz + j0 + (i & 15)] = (__bf16)0.f;
  grid_barrier(cnt, gen, NWG_REC, 1u);

  float ust[2] = {0.f, 0.f};

  for (int t = 0; t < Tsz; ++t) {
    const __bf16* rin = rbuf + (long)(t & 1) * Bsz * Nsz;
    __bf16* rout      = rbuf + (long)((t + 1) & 1) * Bsz * Nsz;

    v8f acc = {};
#pragma unroll
    for (int kk = 0; kk < 8; ++kk) {
      int k = (ks * 8 + kk) * 32;
      v16bf a = load_a_frag(rin, Nsz, m0, k);
      acc = wmma_bf16(a, load_b_frag(sW, Nsz, 0, k), acc);
    }
#pragma unroll
    for (int v = 0; v < 8; ++v)
      sAcc[(wave * 16 + v + ((lane & 16) ? 8 : 0)) * 16 + (lane & 15)] = acc[v];
    __syncthreads();
#pragma unroll
    for (int pi = 0; pi < 2; ++pi) {
      int p = tid + pi * 256;
      int m = p >> 4, n = p & 15;
      int mtl = m >> 4, ml = m & 15;
      float s = 0.f;
#pragma unroll
      for (int w2 = 0; w2 < 4; ++w2)
        s += sAcc[((w2 * 2 + mtl) * 16 + ml) * 16 + n];
      float I = Iext[((long)m * Tsz + t) * Nsz + j0 + n];
      float u = ust[pi];
      u = u + DT_STEP * (-u + s + I);
      ust[pi] = u;
      useq[((long)m * Tsz + t) * Nsz + j0 + n] = (__bf16)u;
      rout[(long)m * Nsz + j0 + n] = (__bf16)fmaxf(u, 0.f);
    }
    grid_barrier(cnt, gen, NWG_REC, (unsigned)(t + 2));
  }
}

// ---------------------------------------------------------------------------
// Projections: q_z_mean / q_z_logvar / q_k_logits + reparameterized z.
// grid = (B*T/32) WGs; wave w: batch-tile w&1, column group w>>1.
// ---------------------------------------------------------------------------
__global__ void proj_kernel(const __bf16* __restrict__ hseq,   // (B*T,H)
                            const __bf16* __restrict__ Wz,
                            const __bf16* __restrict__ Wlv,
                            const __bf16* __restrict__ Wk16,   // padded 16xH
                            const float* __restrict__ bz,
                            const float* __restrict__ blv,
                            const float* __restrict__ bk,
                            const float* __restrict__ eps,     // (B*T,L)
                            float* __restrict__ out_mean, float* __restrict__ out_lv,
                            float* __restrict__ out_logits, float* __restrict__ out_z,
                            __bf16* __restrict__ zbf) {
  const int tid = threadIdx.x, wave = tid >> 5, lane = tid & 31;
  const int mt = wave & 1, g = wave >> 1;          // g in 0..3
  const long row0 = (long)blockIdx.x * 32 + mt * 16;
  const int n0 = g * 16;
  const __bf16* A = hseq + row0 * Hsz;

  v8f am = {}, al = {}, ag = {};
  for (int k = 0; k < Hsz; k += 32) {
    v16bf a = load_a_frag(A, Hsz, 0, k);
    am = wmma_bf16(a, load_b_frag(Wz, Hsz, n0, k), am);
    al = wmma_bf16(a, load_b_frag(Wlv, Hsz, n0, k), al);
    if (g == 0) ag = wmma_bf16(a, load_b_frag(Wk16, Hsz, 0, k), ag);  // wave-uniform
  }
#pragma unroll
  for (int v = 0; v < 8; ++v) {
    long m = row0 + v + ((lane & 16) ? 8 : 0);
    int n = lane & 15;
    float mean = am[v] + bz[n0 + n];
    float lv = al[v] + blv[n0 + n];
    long idx = m * Lsz + n0 + n;
    out_mean[idx] = mean;
    out_lv[idx] = lv;
    float z = mean + eps[idx] * expf(0.5f * lv);
    out_z[idx] = z;
    zbf[idx] = (__bf16)z;
    if (g == 0 && n < Kcat) out_logits[m * Kcat + n] = ag[v] + bk[n];
  }
}

// ---------------------------------------------------------------------------
// Generic WMMA GEMM: C(M,N) = A(bf16, lda) @ W(bf16 N x K, ldb)^T + bias.
// grid (N/128, M/32), 256 threads; wave w: batch-tile w&1, ntiles (w>>1)*2+{0,1}.
// ---------------------------------------------------------------------------
__global__ void gemm_kernel(const __bf16* __restrict__ A, long lda,
                            const __bf16* __restrict__ W, long ldb,
                            const float* __restrict__ bias,
                            float* __restrict__ C, long ldc, int K) {
  const int tid = threadIdx.x, wave = tid >> 5, lane = tid & 31;
  const int mt = wave & 1;
  const int nt = blockIdx.x * 8 + (wave >> 1) * 2;
  const long row0 = (long)blockIdx.y * 32 + mt * 16;
  const __bf16* Ap = A + row0 * lda;

  v8f c0 = {}, c1 = {};
  for (int k = 0; k < K; k += 32) {
    v16bf a = load_a_frag(Ap, lda, 0, k);
    c0 = wmma_bf16(a, load_b_frag(W, ldb, nt * 16, k), c0);
    c1 = wmma_bf16(a, load_b_frag(W, ldb, nt * 16 + 16, k), c1);
  }
#pragma unroll
  for (int v = 0; v < 8; ++v) {
    long m = row0 + v + ((lane & 16) ? 8 : 0);
    int n = lane & 15;
    C[m * ldc + nt * 16 + n] = c0[v] + bias[nt * 16 + n];
    C[m * ldc + nt * 16 + 16 + n] = c1[v] + bias[nt * 16 + 16 + n];
  }
}

// ---------------------------------------------------------------------------
// Gumbel-max categorical sample (first max wins, like jnp.argmax).
// ---------------------------------------------------------------------------
__global__ void ksample_kernel(const float* __restrict__ logits,
                               const float* __restrict__ gu,
                               float* __restrict__ outk, int rows) {
  int r = blockIdx.x * blockDim.x + threadIdx.x;
  if (r >= rows) return;
  float best = -1e30f;
  int bi = 0;
#pragma unroll
  for (int j = 0; j < Kcat; ++j) {
    float g = -logf(-logf(gu[(long)r * Kcat + j]));
    float v = logits[(long)r * Kcat + j] + g;
    if (v > best) { best = v; bi = j; }
  }
  outk[r] = (float)bi;
}

// ---------------------------------------------------------------------------
// Host side
// ---------------------------------------------------------------------------
extern "C" void kernel_launch(void* const* d_in, const int* in_sizes, int n_in,
                              void* d_out, int out_size, void* d_ws, size_t ws_size,
                              hipStream_t stream) {
  (void)in_sizes; (void)n_in; (void)out_size; (void)ws_size;
  const float* y     = (const float*)d_in[0];
  const float* eps   = (const float*)d_in[1];
  const float* gu    = (const float*)d_in[2];
  const float* W_ih  = (const float*)d_in[3];
  const float* W_hh  = (const float*)d_in[4];
  const float* b_ih  = (const float*)d_in[5];
  const float* b_hh  = (const float*)d_in[6];
  const float* Wz    = (const float*)d_in[7];
  const float* bz    = (const float*)d_in[8];
  const float* Wlv   = (const float*)d_in[9];
  const float* blv   = (const float*)d_in[10];
  const float* Wk    = (const float*)d_in[11];
  const float* bk    = (const float*)d_in[12];
  const float* W_int = (const float*)d_in[13];
  const float* b_int = (const float*)d_in[14];
  const float* W_rec = (const float*)d_in[15];
  const float* W_em  = (const float*)d_in[16];
  const float* b_em  = (const float*)d_in[17];

  // Output layout (f32 elements, reference return order)
  float* out = (float*)d_out;
  const long BT = (long)Bsz * Tsz;
  float* o_yrec   = out;                                 // B*T*D
  float* o_mean   = o_yrec + BT * Dsz;                   // B*T*L
  float* o_lv     = o_mean + BT * Lsz;                   // B*T*L
  float* o_logits = o_lv + BT * Lsz;                     // B*T*K
  float* o_z      = o_logits + BT * Kcat;                // B*T*L
  float* o_k      = o_z + BT * Lsz;                      // B*T

  // Workspace carve-up (256B aligned)
  char* ws = (char*)d_ws;
  size_t off = 0;
  auto alloc = [&](size_t bytes) -> char* {
    char* p = ws + off;
    off = (off + bytes + 255) & ~(size_t)255;
    return p;
  };
  __bf16* ybf   = (__bf16*)alloc(BT * Dsz * 2);
  __bf16* Wihb  = (__bf16*)alloc((size_t)4 * Hsz * Dsz * 2);
  __bf16* Whhb  = (__bf16*)alloc((size_t)4 * Hsz * Hsz * 2);
  __bf16* Wzb   = (__bf16*)alloc((size_t)Lsz * Hsz * 2);
  __bf16* Wlvb  = (__bf16*)alloc((size_t)Lsz * Hsz * 2);
  __bf16* Wk16  = (__bf16*)alloc((size_t)16 * Hsz * 2);
  __bf16* Wintb = (__bf16*)alloc((size_t)Nsz * Lsz * 2);
  __bf16* Wrecb = (__bf16*)alloc((size_t)Nsz * Nsz * 2);
  __bf16* Wemb  = (__bf16*)alloc((size_t)Dsz * Nsz * 2);
  __bf16* hbuf  = (__bf16*)alloc((size_t)2 * Bsz * Hsz * 2);
  __bf16* hseq  = (__bf16*)alloc(BT * Hsz * 2);
  __bf16* zbf   = (__bf16*)alloc(BT * Lsz * 2);
  float*  Iext  = (float*)alloc(BT * Nsz * 4);
  __bf16* rbuf  = (__bf16*)alloc((size_t)2 * Bsz * Nsz * 2);
  __bf16* useq  = (__bf16*)alloc(BT * Nsz * 2);
  unsigned* bar = (unsigned*)alloc(256);   // [0,1]=LSTM cnt/gen, [2,3]=CANN

  init_barrier_kernel<<<1, 32, 0, stream>>>(bar);

  auto cvt = [&](const float* s, __bf16* d, long n) {
    f2bf_kernel<<<(unsigned)((n + 255) / 256), 256, 0, stream>>>(s, d, n);
  };
  cvt(y, ybf, BT * Dsz);
  cvt(W_ih, Wihb, (long)4 * Hsz * Dsz);
  cvt(W_hh, Whhb, (long)4 * Hsz * Hsz);
  cvt(Wz, Wzb, (long)Lsz * Hsz);
  cvt(Wlv, Wlvb, (long)Lsz * Hsz);
  cvt(W_int, Wintb, (long)Nsz * Lsz);
  cvt(W_rec, Wrecb, (long)Nsz * Nsz);
  cvt(W_em, Wemb, (long)Dsz * Nsz);
  wk_pad_kernel<<<(16 * Hsz + 255) / 256, 256, 0, stream>>>(Wk, Wk16);

  // --- LSTM recurrence (persistent, W_hh slice in 128KB LDS) ---
  size_t smem_lstm = (size_t)64 * Hsz * 2 + (size_t)64 * Dsz * 2 +
                     (size_t)4 * Bsz * 16 * 4 + 64 * 4;   // 155,904 B < 320KB/WGP
  lstm_scan_kernel<<<NWG_REC, 256, smem_lstm, stream>>>(
      ybf, Wihb, Whhb, b_ih, b_hh, hbuf, hseq, bar + 0, bar + 1);

  // --- VAE projections + reparameterization ---
  proj_kernel<<<(unsigned)(BT / 32), 256, 0, stream>>>(
      hseq, Wzb, Wlvb, Wk16, bz, blv, bk, eps, o_mean, o_lv, o_logits, o_z, zbf);

  // --- interface: I_ext = z @ W_int^T + b_int  (M=BT, N=1024, K=64) ---
  gemm_kernel<<<dim3(Nsz / 128, (unsigned)(BT / 32)), 256, 0, stream>>>(
      zbf, Lsz, Wintb, Lsz, b_int, Iext, Nsz, Lsz);

  // --- categorical sample ---
  ksample_kernel<<<(unsigned)((BT + 255) / 256), 256, 0, stream>>>(
      o_logits, gu, o_k, (int)BT);

  // --- CANN recurrence (persistent, W_rec slice in 32KB LDS) ---
  size_t smem_cann = (size_t)16 * Nsz * 2 + (size_t)8 * 16 * 16 * 4;  // 40,960 B
  cann_scan_kernel<<<NWG_REC, 256, smem_cann, stream>>>(
      Wrecb, Iext, rbuf, useq, bar + 2, bar + 3);

  // --- emission: y_recon = u_seq @ W_em^T + b_em  (M=BT, N=128, K=1024) ---
  gemm_kernel<<<dim3(Dsz / 128, (unsigned)(BT / 32)), 256, 0, stream>>>(
      useq, Nsz, Wemb, Nsz, b_em, o_yrec, Dsz, Hsz);
}
